// Qwen3_5DecoderLayer_56650618635073
// MI455X (gfx1250) — matrix operand order repaired and checked by reading, not compiled, not run on hardware
//
#include <hip/hip_runtime.h>
#include <hip/hip_bf16.h>
#include <math.h>

// ---------------------------------------------------------------------------
// Problem constants (match reference)
// ---------------------------------------------------------------------------
static constexpr int Bc   = 4;
static constexpr int Sc   = 2048;
static constexpr int Hc   = 1024;
static constexpr int NHc  = 8;
static constexpr int NKVc = 2;
static constexpr int HDc  = 256;
static constexpr int ROTc = 64;
static constexpr int FFc  = 3584;
static constexpr int Mc   = Bc * Sc;          // 8192 rows
static constexpr float EPSc = 1e-6f;

typedef __bf16 bf16;
typedef bf16  v16bf __attribute__((ext_vector_type(16)));
typedef float v8f   __attribute__((ext_vector_type(8)));

union Frag { v16bf v; unsigned u[8]; };

#define DEVINL __device__ __forceinline__

DEVINL unsigned short f2bf(float f) {
  unsigned u = __float_as_uint(f);
  unsigned r = u + 0x7FFFu + ((u >> 16) & 1u);   // round-to-nearest-even
  return (unsigned short)(r >> 16);
}

// ---------------------------------------------------------------------------
// CDNA5 async memory->LDS copy (ASYNCcnt-tracked) + waits.
// ldsoff: byte offset in workgroup LDS (low 32 bits of an AS(3) pointer).
// gaddr : 64-bit global address (VGPR pair).
// ---------------------------------------------------------------------------
DEVINL void async_copy_b128(unsigned ldsoff, const void* gaddr) {
  asm volatile("global_load_async_to_lds_b128 %0, %1, off"
               :: "v"(ldsoff), "v"(gaddr)
               : "memory");
}
template <int N>
DEVINL void wait_asynccnt() {
  asm volatile("s_wait_asynccnt %0" :: "i"(N) : "memory");
}

// ---------------------------------------------------------------------------
// fp32 -> bf16 cast (weights)
// ---------------------------------------------------------------------------
__global__ __launch_bounds__(256)
void cast_bf16_kernel(const float* __restrict__ X, unsigned short* __restrict__ Y, size_t n) {
  size_t i = (size_t)blockIdx.x * 256 + threadIdx.x;
  if (i < n) Y[i] = f2bf(X[i]);
}

// ---------------------------------------------------------------------------
// Row RMSNorm (H elements per row) -> bf16
// ---------------------------------------------------------------------------
__global__ __launch_bounds__(256)
void rmsnorm_rows_kernel(const float* __restrict__ X, const float* __restrict__ W,
                         unsigned short* __restrict__ Y, int Hdim) {
  __shared__ float red[8];
  const int row = blockIdx.x;
  const float* x = X + (size_t)row * Hdim;
  float s = 0.f;
  for (int i = threadIdx.x; i < Hdim; i += 256) { float v = x[i]; s += v * v; }
#pragma unroll
  for (int m = 16; m >= 1; m >>= 1) s += __shfl_xor(s, m, 32);
  if ((threadIdx.x & 31) == 0) red[threadIdx.x >> 5] = s;
  __syncthreads();
  if (threadIdx.x == 0) {
    float t = 0.f;
    for (int i = 0; i < 8; ++i) t += red[i];
    red[0] = rsqrtf(t / (float)Hdim + EPSc);
  }
  __syncthreads();
  const float rms = red[0];
  unsigned short* y = Y + (size_t)row * Hdim;
  for (int i = threadIdx.x; i < Hdim; i += 256) y[i] = f2bf(x[i] * rms * W[i]);
}

// ---------------------------------------------------------------------------
// Tiled bf16 WMMA GEMM:  C[M,N] (+Res) = A[M,K] @ B[K,N]
// Block tile 128x128, K-step 32, 8 waves, wave tile 32x64 (2x4 WMMA accums).
// Double-buffered LDS staged with GLOBAL_LOAD_ASYNC_TO_LDS_B128:
// tile t+1 streams into the other buffer while WMMAs consume tile t; each
// stage is 4 async instructions per wave, so s_wait_asynccnt<=4 lets the
// next-tile copies stay in flight.
// EPI: 0 -> C = A@B ; 1 -> C = Res + A@B
// ---------------------------------------------------------------------------
template<int EPI>
__global__ __launch_bounds__(256)
void gemm_bf16_kernel(const unsigned short* __restrict__ A,
                      const unsigned short* __restrict__ Bw,
                      const float* __restrict__ Res,
                      float* __restrict__ C,
                      int M, int N, int K) {
  __shared__ __align__(16) unsigned short lsA[2][128 * 32];
  __shared__ __align__(16) unsigned short lsB[2][32 * 128];

  const int t    = threadIdx.x;
  const int lane = t & 31;
  const int w    = t >> 5;
  const int wm   = w >> 1;                 // 0..3
  const int wn   = w & 1;                  // 0..1
  const int m0   = blockIdx.y * 128;
  const int n0   = blockIdx.x * 128;

  v8f acc[2][4];
#pragma unroll
  for (int i = 0; i < 2; ++i)
#pragma unroll
    for (int j = 0; j < 4; ++j)
#pragma unroll
      for (int r = 0; r < 8; ++r) acc[i][j][r] = 0.f;

  const int mr = lane & 15;                // A-fragment row within tile
  const int kb = (lane >> 4) << 3;         // K half-offset (0 or 8)

  // Issue one stage of async memory->LDS copies (4 x b128 per thread-slot).
  auto stage = [&](int buf, int k0) {
    // A tile (128x32, row-major): 2 x 16B per thread
    int r   = t >> 2;
    int seg = (t & 3) << 3;
#pragma unroll
    for (int rr = 0; rr < 2; ++rr, r += 64)
      async_copy_b128((unsigned)(size_t)(void*)&lsA[buf][r * 32 + seg],
                      &A[(size_t)(m0 + r) * K + k0 + seg]);
    // B tile (32x128, row-major KxN): 2 x 16B per thread
    int k  = t >> 3;
    int nb = (t & 7) << 4;
    const unsigned short* src = &Bw[(size_t)(k0 + k) * N + n0 + nb];
    unsigned short* dst = &lsB[buf][k * 128 + nb];
    async_copy_b128((unsigned)(size_t)(void*)dst, src);
    async_copy_b128((unsigned)(size_t)(void*)(dst + 8), src + 8);
  };

  const int nit = K >> 5;                  // K / 32
  stage(0, 0);

  for (int it = 0; it < nit; ++it) {
    const int cur = it & 1;
    if (it + 1 < nit) {
      stage(cur ^ 1, (it + 1) << 5);
      wait_asynccnt<4>();                  // current tile done; next in flight
    } else {
      wait_asynccnt<0>();
    }
    __syncthreads();                       // current buffer visible to all waves

    // A fragments: lane holds row mr, K-pairs per ISA 16-bit A layout
    Frag af[2];
#pragma unroll
    for (int rt = 0; rt < 2; ++rt) {
      const unsigned short* pa = &lsA[cur][(wm * 32 + rt * 16 + mr) * 32];
#pragma unroll
      for (int i = 0; i < 8; ++i) {
        int k = (i < 4) ? (kb + 2 * i) : (16 + kb + 2 * (i - 4));
        af[rt].u[i] = *(const unsigned*)(pa + k);
      }
    }
    // B fragments: lane = N column (lane&15), VGPR halves = K pairs (table order, strided in lsB)
#pragma unroll
    for (int ct = 0; ct < 4; ++ct) {
      Frag bfr;
      const unsigned short* pb = &lsB[cur][wn * 64 + ct * 16 + mr];
#pragma unroll
      for (int i = 0; i < 8; ++i) {
        int k = (i < 4) ? (kb + 2 * i) : (16 + kb + 2 * (i - 4));
        bfr.u[i] = (unsigned)pb[k * 128] | ((unsigned)pb[(k + 1) * 128] << 16);
      }
#pragma unroll
      for (int rt = 0; rt < 2; ++rt)
        acc[rt][ct] = __builtin_amdgcn_wmma_f32_16x16x32_bf16(
            false, af[rt].v, false, bfr.v, (short)0, acc[rt][ct], false, false);
    }
    __syncthreads();                       // all reads done before overwrite
  }

  // Epilogue: C layout -> VGPR r: lanes 0-15 M=r, lanes 16-31 M=r+8; N=lane&15
  const int rowo = (lane >> 4) << 3;
  const int colo = lane & 15;
#pragma unroll
  for (int rt = 0; rt < 2; ++rt)
#pragma unroll
    for (int ct = 0; ct < 4; ++ct)
#pragma unroll
      for (int r = 0; r < 8; ++r) {
        int row = m0 + wm * 32 + rt * 16 + r + rowo;
        int col = n0 + wn * 64 + ct * 16 + colo;
        float v = acc[rt][ct][r];
        if (EPI == 1) v += Res[(size_t)row * N + col];
        C[(size_t)row * N + col] = v;
      }
}

// ---------------------------------------------------------------------------
// Q pre-process: per (b,s,head): rmsnorm over HD=256 + RoPE on first 64 dims.
// Input QG fp32 (B,S,NH,2*HD); output Q bf16 (B,NH,S,HD)
// ---------------------------------------------------------------------------
__global__ __launch_bounds__(256)
void q_pre_kernel(const float* __restrict__ QG, const float* __restrict__ Cos,
                  const float* __restrict__ Sin, const float* __restrict__ NW,
                  unsigned short* __restrict__ Qo) {
  __shared__ float red[8];
  __shared__ float sh[256];
  const int row = blockIdx.x;            // b*S + s
  const int h   = blockIdx.y;
  const int hd  = threadIdx.x;
  const int b   = row / Sc;
  const int s   = row - b * Sc;
  float v  = QG[((size_t)row * NHc + h) * (2 * HDc) + hd];
  float sq = v * v;
#pragma unroll
  for (int m = 16; m >= 1; m >>= 1) sq += __shfl_xor(sq, m, 32);
  if ((hd & 31) == 0) red[hd >> 5] = sq;
  __syncthreads();
  if (hd == 0) {
    float t = 0.f;
    for (int i = 0; i < 8; ++i) t += red[i];
    red[0] = rsqrtf(t / (float)HDc + EPSc);
  }
  __syncthreads();
  float nv = v * red[0] * NW[hd];
  sh[hd] = nv;
  __syncthreads();
  float out = nv;
  if (hd < ROTc) {
    float c  = Cos[(size_t)row * ROTc + hd];
    float sn = Sin[(size_t)row * ROTc + hd];
    float rot = (hd < ROTc / 2) ? -sh[hd + ROTc / 2] : sh[hd - ROTc / 2];
    out = nv * c + rot * sn;
  }
  Qo[(((size_t)b * NHc + h) * Sc + s) * HDc + hd] = f2bf(out);
}

// K pre-process: rmsnorm + RoPE; output TRANSPOSED bf16 (B,NKV,HD,S)
__global__ __launch_bounds__(256)
void k_pre_kernel(const float* __restrict__ Kf, const float* __restrict__ Cos,
                  const float* __restrict__ Sin, const float* __restrict__ NW,
                  unsigned short* __restrict__ KTo) {
  __shared__ float red[8];
  __shared__ float sh[256];
  const int row = blockIdx.x;            // b*S + s
  const int kv  = blockIdx.y;
  const int hd  = threadIdx.x;
  const int b   = row / Sc;
  const int s   = row - b * Sc;
  float v  = Kf[((size_t)row * NKVc + kv) * HDc + hd];
  float sq = v * v;
#pragma unroll
  for (int m = 16; m >= 1; m >>= 1) sq += __shfl_xor(sq, m, 32);
  if ((hd & 31) == 0) red[hd >> 5] = sq;
  __syncthreads();
  if (hd == 0) {
    float t = 0.f;
    for (int i = 0; i < 8; ++i) t += red[i];
    red[0] = rsqrtf(t / (float)HDc + EPSc);
  }
  __syncthreads();
  float nv = v * red[0] * NW[hd];
  sh[hd] = nv;
  __syncthreads();
  float out = nv;
  if (hd < ROTc) {
    float c  = Cos[(size_t)row * ROTc + hd];
    float sn = Sin[(size_t)row * ROTc + hd];
    float rot = (hd < ROTc / 2) ? -sh[hd + ROTc / 2] : sh[hd - ROTc / 2];
    out = nv * c + rot * sn;
  }
  KTo[(((size_t)b * NKVc + kv) * HDc + hd) * Sc + s] = f2bf(out);
}

// V: fp32 (B,S,NKV,HD) -> bf16 (B,NKV,S,HD)
__global__ __launch_bounds__(256)
void v_pre_kernel(const float* __restrict__ Vf, unsigned short* __restrict__ Vo) {
  int idx = blockIdx.x * 256 + threadIdx.x;     // B*S*NKV*HD = 4,194,304
  int hd = idx & (HDc - 1);
  int r  = idx >> 8;
  int kv = r & (NKVc - 1);
  r >>= 1;
  int s = r & (Sc - 1);
  int b = r >> 11;
  Vo[(((size_t)b * NKVc + kv) * Sc + s) * HDc + hd] = f2bf(Vf[idx]);
}

// ---------------------------------------------------------------------------
// Flash attention, causal, GQA. One wave handles 16 query rows of one head.
//   Q  : bf16 (B,NH,S,HD)          (A-fragments)
//   KT : bf16 (B,NKV,HD,S)         (B-fragments for Q@K^T, contiguous per lane)
//   V  : bf16 (B,NKV,S,HD)         (B-fragments for P@V, contiguous per lane)
//   QG : fp32 (B,S,NH,2*HD)        (gate = second half)
//   OUT: bf16 (B,S,NH*HD)          attn * sigmoid(gate)
// ---------------------------------------------------------------------------
__global__ __launch_bounds__(32)
void flash_attn_kernel(const unsigned short* __restrict__ Q,
                       const unsigned short* __restrict__ KT,
                       const unsigned short* __restrict__ V,
                       const float* __restrict__ QG,
                       unsigned short* __restrict__ OUT) {
  __shared__ __align__(16) unsigned short lsP[16 * 32];

  const int lane = threadIdx.x & 31;
  const int qi   = blockIdx.x * 16;
  const int h    = blockIdx.y;
  const int b    = blockIdx.z;
  const int kv   = h / (NHc / NKVc);

  const unsigned short* Qb  = Q  + (((size_t)b * NHc  + h ) * Sc) * HDc;
  const unsigned short* KTb = KT + (((size_t)b * NKVc + kv) * HDc) * Sc;
  const unsigned short* Vb  = V  + (((size_t)b * NKVc + kv) * Sc) * HDc;

  const int mr   = lane & 15;
  const int kb   = (lane >> 4) << 3;
  const int rowo = (lane >> 4) << 3;
  const int colo = lane & 15;
  const float scale = 0.0625f;             // 1/sqrt(256)
  const float NEG = -3.0e38f;

  // resident Q fragments: 8 K-chunks of 32 covering HD=256
  Frag qf[8];
  {
    const unsigned short* pq = Qb + (size_t)(qi + mr) * HDc;
#pragma unroll
    for (int hc = 0; hc < 8; ++hc)
#pragma unroll
      for (int i = 0; i < 8; ++i) {
        int k = hc * 32 + ((i < 4) ? (kb + 2 * i) : (16 + kb + 2 * (i - 4)));
        qf[hc].u[i] = *(const unsigned*)(pq + k);
      }
  }

  v8f o[16];
#pragma unroll
  for (int j = 0; j < 16; ++j)
#pragma unroll
    for (int r = 0; r < 8; ++r) o[j][r] = 0.f;
  float mrow[8], lrow[8], cfac[8];
#pragma unroll
  for (int r = 0; r < 8; ++r) { mrow[r] = NEG; lrow[r] = 0.f; }

  for (int kj = 0; kj < qi + 16; kj += 32) {
    // ---- scores: S[16 x 32] = Q @ K^T, two 16x16 accumulators ----
    v8f s0, s1;
#pragma unroll
    for (int r = 0; r < 8; ++r) { s0[r] = 0.f; s1[r] = 0.f; }
#pragma unroll
    for (int hc = 0; hc < 8; ++hc) {
      const unsigned short* pk = KTb + (size_t)(hc * 32) * Sc + kj + colo;
      Frag b0, b1;
#pragma unroll
      for (int i = 0; i < 8; ++i) {
        int k = (i < 4) ? (kb + 2 * i) : (16 + kb + 2 * (i - 4));
        b0.u[i] = (unsigned)pk[(size_t)k * Sc] | ((unsigned)pk[(size_t)(k + 1) * Sc] << 16);
        b1.u[i] = (unsigned)pk[(size_t)k * Sc + 16] | ((unsigned)pk[(size_t)(k + 1) * Sc + 16] << 16);
      }
      s0 = __builtin_amdgcn_wmma_f32_16x16x32_bf16(false, qf[hc].v, false, b0.v,
                                                   (short)0, s0, false, false);
      s1 = __builtin_amdgcn_wmma_f32_16x16x32_bf16(false, qf[hc].v, false, b1.v,
                                                   (short)0, s1, false, false);
    }

    const bool need_mask = (kj + 31) > qi;
    // ---- online softmax (rows split r / r+8 across lane halves) ----
#pragma unroll
    for (int r = 0; r < 8; ++r) {
      int row = qi + r + rowo;
      float a0 = s0[r] * scale;
      float a1 = s1[r] * scale;
      if (need_mask) {
        if (kj + colo      > row) a0 = NEG;
        if (kj + 16 + colo > row) a1 = NEG;
      }
      float mx = fmaxf(a0, a1);
      mx = fmaxf(mx, __shfl_xor(mx, 1, 16));
      mx = fmaxf(mx, __shfl_xor(mx, 2, 16));
      mx = fmaxf(mx, __shfl_xor(mx, 4, 16));
      mx = fmaxf(mx, __shfl_xor(mx, 8, 16));
      float mn = fmaxf(mrow[r], mx);
      float p0 = __expf(a0 - mn);
      float p1 = __expf(a1 - mn);
      s0[r] = p0; s1[r] = p1;
      float rs = p0 + p1;
      rs += __shfl_xor(rs, 1, 16);
      rs += __shfl_xor(rs, 2, 16);
      rs += __shfl_xor(rs, 4, 16);
      rs += __shfl_xor(rs, 8, 16);
      float cf = __expf(mrow[r] - mn);
      lrow[r] = lrow[r] * cf + rs;
      mrow[r] = mn;
      cfac[r] = cf;
    }

    // ---- stage P (16x32) to LDS, reload as A-fragment ----
#pragma unroll
    for (int r = 0; r < 8; ++r) {
      int rl = r + rowo;
      lsP[rl * 32 + colo]      = f2bf(s0[r]);
      lsP[rl * 32 + 16 + colo] = f2bf(s1[r]);
    }
    __syncthreads();
    Frag pf;
    {
      const unsigned short* pp = &lsP[mr * 32];
#pragma unroll
      for (int i = 0; i < 8; ++i) {
        int k = (i < 4) ? (kb + 2 * i) : (16 + kb + 2 * (i - 4));
        pf.u[i] = *(const unsigned*)(pp + k);
      }
    }
    __syncthreads();

    // ---- rescale O, accumulate O += P @ V over all 16 hd-chunks ----
#pragma unroll
    for (int j = 0; j < 16; ++j)
#pragma unroll
      for (int r = 0; r < 8; ++r) o[j][r] *= cfac[r];
#pragma unroll
    for (int j = 0; j < 16; ++j) {
      const unsigned short* pv = Vb + (size_t)kj * HDc + j * 16 + colo;
      Frag vf;
#pragma unroll
      for (int i = 0; i < 8; ++i) {
        int k = (i < 4) ? (kb + 2 * i) : (16 + kb + 2 * (i - 4));
        vf.u[i] = (unsigned)pv[(size_t)k * HDc] | ((unsigned)pv[(size_t)(k + 1) * HDc] << 16);
      }
      o[j] = __builtin_amdgcn_wmma_f32_16x16x32_bf16(false, pf.v, false, vf.v,
                                                     (short)0, o[j], false, false);
    }
  }

  // ---- epilogue: normalize, gate with sigmoid, store bf16 ----
  float invl[8];
#pragma unroll
  for (int r = 0; r < 8; ++r) invl[r] = 1.f / lrow[r];
#pragma unroll
  for (int j = 0; j < 16; ++j)
#pragma unroll
    for (int r = 0; r < 8; ++r) {
      int row = qi + r + rowo;
      int col = j * 16 + colo;
      float val = o[j][r] * invl[r];
      float g = QG[(((size_t)b * Sc + row) * NHc + h) * (2 * HDc) + HDc + col];
      val *= 1.f / (1.f + __expf(-g));
      OUT[(((size_t)b * Sc + row) * NHc + h) * HDc + col] = f2bf(val);
    }
}

// ---------------------------------------------------------------------------
// SiLU(gate) * up -> bf16
// ---------------------------------------------------------------------------
__global__ __launch_bounds__(256)
void silu_mul_kernel(const float* __restrict__ G, const float* __restrict__ U,
                     unsigned short* __restrict__ A, size_t n) {
  size_t i = (size_t)blockIdx.x * 256 + threadIdx.x;
  if (i < n) {
    float g = G[i];
    float sg = g / (1.f + __expf(-g));
    A[i] = f2bf(sg * U[i]);
  }
}

// ---------------------------------------------------------------------------
// Host-side orchestration
// ---------------------------------------------------------------------------
extern "C" void kernel_launch(void* const* d_in, const int* in_sizes, int n_in,
                              void* d_out, int out_size, void* d_ws, size_t ws_size,
                              hipStream_t stream) {
  (void)in_sizes; (void)n_in; (void)out_size; (void)ws_size;
  const float* hidden = (const float*)d_in[0];
  const float* cosb   = (const float*)d_in[1];
  const float* sinb   = (const float*)d_in[2];
  const float* Wq     = (const float*)d_in[3];
  const float* Wk     = (const float*)d_in[4];
  const float* Wv     = (const float*)d_in[5];
  const float* Wo     = (const float*)d_in[6];
  const float* qnw    = (const float*)d_in[7];
  const float* knw    = (const float*)d_in[8];
  const float* ln1    = (const float*)d_in[9];
  const float* ln2    = (const float*)d_in[10];
  const float* Wg     = (const float*)d_in[11];
  const float* Wu     = (const float*)d_in[12];
  const float* Wd     = (const float*)d_in[13];
  float* out = (float*)d_out;

  char* p = (char*)d_ws;
  auto alloc = [&](size_t bytes) -> void* {
    void* r = (void*)p;
    p += (bytes + 255) & ~(size_t)255;
    return r;
  };

  const int NQ = NHc * HDc * 2;          // 4096
  const int NKVD = NKVc * HDc;           // 512

  unsigned short* x_bf   = (unsigned short*)alloc((size_t)Mc * Hc * 2);
  unsigned short* wq_bf  = (unsigned short*)alloc((size_t)Hc * NQ * 2);
  unsigned short* wk_bf  = (unsigned short*)alloc((size_t)Hc * NKVD * 2);
  unsigned short* wv_bf  = (unsigned short*)alloc((size_t)Hc * NKVD * 2);
  unsigned short* wo_bf  = (unsigned short*)alloc((size_t)(NHc * HDc) * Hc * 2);
  unsigned short* wg_bf  = (unsigned short*)alloc((size_t)Hc * FFc * 2);
  unsigned short* wu_bf  = (unsigned short*)alloc((size_t)Hc * FFc * 2);
  unsigned short* wd_bf  = (unsigned short*)alloc((size_t)FFc * Hc * 2);
  float*          qg_f   = (float*)alloc((size_t)Mc * NQ * 4);
  float*          k_f    = (float*)alloc((size_t)Mc * NKVD * 4);
  float*          v_f    = (float*)alloc((size_t)Mc * NKVD * 4);
  unsigned short* q_bf   = (unsigned short*)alloc((size_t)Bc * NHc * Sc * HDc * 2);
  unsigned short* kt_bf  = (unsigned short*)alloc((size_t)Bc * NKVc * HDc * Sc * 2);
  unsigned short* v_bf   = (unsigned short*)alloc((size_t)Bc * NKVc * Sc * HDc * 2);
  unsigned short* at_bf  = (unsigned short*)alloc((size_t)Mc * (NHc * HDc) * 2);
  float*          h_f    = (float*)alloc((size_t)Mc * Hc * 4);
  unsigned short* y_bf   = (unsigned short*)alloc((size_t)Mc * Hc * 2);
  float*          g_f    = (float*)alloc((size_t)Mc * FFc * 4);
  float*          u_f    = (float*)alloc((size_t)Mc * FFc * 4);
  unsigned short* act_bf = (unsigned short*)alloc((size_t)Mc * FFc * 2);

  auto cast = [&](const float* src, unsigned short* dst, size_t n) {
    cast_bf16_kernel<<<(unsigned)((n + 255) / 256), 256, 0, stream>>>(src, dst, n);
  };
  cast(Wq, wq_bf, (size_t)Hc * NQ);
  cast(Wk, wk_bf, (size_t)Hc * NKVD);
  cast(Wv, wv_bf, (size_t)Hc * NKVD);
  cast(Wo, wo_bf, (size_t)(NHc * HDc) * Hc);
  cast(Wg, wg_bf, (size_t)Hc * FFc);
  cast(Wu, wu_bf, (size_t)Hc * FFc);
  cast(Wd, wd_bf, (size_t)FFc * Hc);

  // x = rmsnorm(hidden, ln1) -> bf16
  rmsnorm_rows_kernel<<<Mc, 256, 0, stream>>>(hidden, ln1, x_bf, Hc);

  // QKV projections (WMMA GEMMs, async-LDS staged)
  gemm_bf16_kernel<0><<<dim3(NQ / 128, Mc / 128), 256, 0, stream>>>(
      x_bf, wq_bf, nullptr, qg_f, Mc, NQ, Hc);
  gemm_bf16_kernel<0><<<dim3(NKVD / 128, Mc / 128), 256, 0, stream>>>(
      x_bf, wk_bf, nullptr, k_f, Mc, NKVD, Hc);
  gemm_bf16_kernel<0><<<dim3(NKVD / 128, Mc / 128), 256, 0, stream>>>(
      x_bf, wv_bf, nullptr, v_f, Mc, NKVD, Hc);

  // per-head rmsnorm + RoPE, layout shuffles for attention
  q_pre_kernel<<<dim3(Mc, NHc), 256, 0, stream>>>(qg_f, cosb, sinb, qnw, q_bf);
  k_pre_kernel<<<dim3(Mc, NKVc), 256, 0, stream>>>(k_f, cosb, sinb, knw, kt_bf);
  v_pre_kernel<<<(Bc * NKVc * Sc * HDc) / 256, 256, 0, stream>>>(v_f, v_bf);

  // flash attention + sigmoid gating
  flash_attn_kernel<<<dim3(Sc / 16, NHc, Bc), 32, 0, stream>>>(
      q_bf, kt_bf, v_bf, qg_f, at_bf);

  // h = residual + attn @ Wo
  gemm_bf16_kernel<1><<<dim3(Hc / 128, Mc / 128), 256, 0, stream>>>(
      at_bf, wo_bf, hidden, h_f, Mc, Hc, NHc * HDc);

  // y = rmsnorm(h, ln2) -> bf16
  rmsnorm_rows_kernel<<<Mc, 256, 0, stream>>>(h_f, ln2, y_bf, Hc);

  // MLP
  gemm_bf16_kernel<0><<<dim3(FFc / 128, Mc / 128), 256, 0, stream>>>(
      y_bf, wg_bf, nullptr, g_f, Mc, FFc, Hc);
  gemm_bf16_kernel<0><<<dim3(FFc / 128, Mc / 128), 256, 0, stream>>>(
      y_bf, wu_bf, nullptr, u_f, Mc, FFc, Hc);
  silu_mul_kernel<<<(unsigned)(((size_t)Mc * FFc + 255) / 256), 256, 0, stream>>>(
      g_f, u_f, act_bf, (size_t)Mc * FFc);

  // out = h + act @ Wdown
  gemm_bf16_kernel<1><<<dim3(Hc / 128, Mc / 128), 256, 0, stream>>>(
      act_bf, wd_bf, h_f, out, Mc, Hc, FFc);
}
